// BiMamba2Layer_58162447122717
// MI455X (gfx1250) — matrix operand order repaired and checked by reading, not compiled, not run on hardware
//
#include <hip/hip_runtime.h>
#include <hip/hip_bf16.h>
#include <math.h>

#define D_MODEL   512
#define D_STATE   16
#define D_CONV    5
#define HEADDIM   64
#define D_INNER   1024
#define NHEADS    16
#define CONV_DIM  1056
#define D_IN_PROJ 2096
#define B_SZ      16
#define SEQ       960
#define MROWS     (B_SZ * SEQ)
#define EPS       1e-5f

typedef __attribute__((ext_vector_type(16))) __bf16 bf16x16;
typedef __attribute__((ext_vector_type(8)))  __bf16 bf16x8;
typedef __attribute__((ext_vector_type(8)))  float  f32x8;

__device__ __forceinline__ unsigned short f32_to_bf16_rne(float f) {
    unsigned int u = __float_as_uint(f);
    unsigned int r = u + 0x7FFFu + ((u >> 16) & 1u);
    if ((u & 0x7F800000u) == 0x7F800000u) r = u;   // inf/NaN: truncate
    return (unsigned short)(r >> 16);
}

// ---------------- elementwise f32 -> bf16 ----------------
__global__ void cvt_f32_bf16(const float* __restrict__ src,
                             unsigned short* __restrict__ dst, int n) {
    int i = blockIdx.x * blockDim.x + threadIdx.x;
    if (i < n) dst[i] = f32_to_bf16_rne(src[i]);
}

// ---------------- W[K][N] f32 -> Wt[N][K] bf16 ----------------
__global__ void transpose_cvt(const float* __restrict__ W,
                              unsigned short* __restrict__ Wt, int K, int N) {
    int i = blockIdx.x * blockDim.x + threadIdx.x;
    if (i >= N * K) return;
    int n = i / K, k = i % K;
    Wt[i] = f32_to_bf16_rne(W[(size_t)k * N + n]);
}

// ---------------- bf16 WMMA GEMM: C[M][N](f32) = A[M][K](bf16) * Bt[N][K]^T ----
// One wave per block; each wave owns a 16x64 output tile (4 WMMA accumulators).
// Ragged N edge handled by clamping inactive tiles onto the last valid tile
// (loads stay in-bounds, redundant WMMAs are harmless) and guarding stores only,
// so the K-loop is completely branch-free.
// revA: read A rows time-reversed per batch (backward in_proj).
// revC: write C rows time-reversed per batch (backward out_proj).
__global__ __launch_bounds__(32)
void gemm_bf16(const unsigned short* __restrict__ Au,
               const unsigned short* __restrict__ Btu,
               float* __restrict__ C,
               int N, int K, int ldc, int colOff,
               int revA, int revC, int Lseq)
{
    const int lane  = threadIdx.x & 31;
    const int lmod  = lane & 15;
    const int lhalf = (lane >> 4) & 1;
    const int m0 = blockIdx.x * 16;
    const int n0 = blockIdx.y * 64;

    int mA = m0 + lmod;
    if (revA) { int b = mA / Lseq, t = mA % Lseq; mA = b * Lseq + (Lseq - 1 - t); }
    const __bf16* __restrict__ Arow =
        reinterpret_cast<const __bf16*>(Au) + (size_t)mA * K;

    bool act[4];
    const __bf16* __restrict__ Bcol[4];
#pragma unroll
    for (int j = 0; j < 4; ++j) {
        int nt = n0 + j * 16;
        act[j] = (nt < N);
        int ntc = act[j] ? nt : (N - 16);          // clamp: keep loads in-bounds
        Bcol[j] = reinterpret_cast<const __bf16*>(Btu) + (size_t)(ntc + lmod) * K;
    }

    union AF { bf16x16 v; bf16x8 h[2]; };
    const int kaA = lhalf * 16;   // A: lanes 0-15 K 0..15        ; lanes 16-31 K 16..31
    const int kaB = lhalf * 16;   // B: lanes 0-15 K 0..15        ; lanes 16-31 K 16..31

    f32x8 acc0 = {}, acc1 = {}, acc2 = {}, acc3 = {};

    for (int kb = 0; kb < K; kb += 32) {
        AF a, b0, b1, b2, b3;
        a.h[0]  = *reinterpret_cast<const bf16x8*>(Arow + kb + kaA);
        a.h[1]  = *reinterpret_cast<const bf16x8*>(Arow + kb + kaA + 8);
        b0.h[0] = *reinterpret_cast<const bf16x8*>(Bcol[0] + kb + kaB);
        b0.h[1] = *reinterpret_cast<const bf16x8*>(Bcol[0] + kb + kaB + 8);
        b1.h[0] = *reinterpret_cast<const bf16x8*>(Bcol[1] + kb + kaB);
        b1.h[1] = *reinterpret_cast<const bf16x8*>(Bcol[1] + kb + kaB + 8);
        b2.h[0] = *reinterpret_cast<const bf16x8*>(Bcol[2] + kb + kaB);
        b2.h[1] = *reinterpret_cast<const bf16x8*>(Bcol[2] + kb + kaB + 8);
        b3.h[0] = *reinterpret_cast<const bf16x8*>(Bcol[3] + kb + kaB);
        b3.h[1] = *reinterpret_cast<const bf16x8*>(Bcol[3] + kb + kaB + 8);
        acc0 = __builtin_amdgcn_wmma_f32_16x16x32_bf16(false, a.v, false, b0.v, (short)0, acc0, false, false);
        acc1 = __builtin_amdgcn_wmma_f32_16x16x32_bf16(false, a.v, false, b1.v, (short)0, acc1, false, false);
        acc2 = __builtin_amdgcn_wmma_f32_16x16x32_bf16(false, a.v, false, b2.v, (short)0, acc2, false, false);
        acc3 = __builtin_amdgcn_wmma_f32_16x16x32_bf16(false, a.v, false, b3.v, (short)0, acc3, false, false);
    }

    f32x8 accv[4] = {acc0, acc1, acc2, acc3};
#pragma unroll
    for (int j = 0; j < 4; ++j) {
        if (!act[j]) continue;
        int n = n0 + j * 16 + lmod;
#pragma unroll
        for (int r = 0; r < 8; ++r) {
            int mo = m0 + lhalf * 8 + r;   // C/D layout: lanes 0-15 -> M=r, lanes 16-31 -> M=8+r
            if (revC) { int b = mo / Lseq, t = mo % Lseq; mo = b * Lseq + (Lseq - 1 - t); }
            C[(size_t)mo * ldc + colOff + n] = accv[j][r];
        }
    }
}

// ---------------- causal depthwise conv (5-tap) + bias + SiLU ----------------
__global__ void conv_silu(const float* __restrict__ zx, const float* __restrict__ cw,
                          const float* __restrict__ cb, float* __restrict__ xc)
{
    int i = blockIdx.x * blockDim.x + threadIdx.x;
    if (i >= MROWS * CONV_DIM) return;
    int m = i / CONV_DIM, c = i % CONV_DIM;
    int b = m / SEQ, t = m % SEQ;
    float acc = cb[c];
#pragma unroll
    for (int j = 0; j < D_CONV; ++j) {
        int ts = t - (D_CONV - 1) + j;
        if (ts >= 0)
            acc += cw[c * D_CONV + j] *
                   zx[((size_t)(b * SEQ + ts)) * D_IN_PROJ + D_INNER + c];
    }
    xc[i] = acc / (1.0f + __expf(-acc));
}

// ---------------- dt = softplus(raw + bias);  dA = exp(-exp(A_log)*dt) --------
__global__ void dt_da(const float* __restrict__ zx, const float* __restrict__ dt_bias,
                      const float* __restrict__ A_log,
                      float* __restrict__ dt, float* __restrict__ dA)
{
    int i = blockIdx.x * blockDim.x + threadIdx.x;
    if (i >= MROWS * NHEADS) return;
    int h = i % NHEADS;
    size_t m = (size_t)(i / NHEADS);
    float xv = zx[m * D_IN_PROJ + D_INNER + CONV_DIM + h] + dt_bias[h];
    float sp = (xv > 20.0f) ? xv : log1pf(__expf(xv));
    dt[i] = sp;
    dA[i] = __expf(-__expf(A_log[h]) * sp);
}

// ---------------- selective scan: block per (batch, head), lane per headdim --
__global__ __launch_bounds__(64)
void ssm_scan(const float* __restrict__ xc, const float* __restrict__ dt,
              const float* __restrict__ dA, const float* __restrict__ Dp,
              float* __restrict__ ys)
{
    int bh = blockIdx.x;
    int b = bh / NHEADS, h = bh % NHEADS;
    int p = threadIdx.x;                 // 0..63
    float hs[D_STATE];
#pragma unroll
    for (int n = 0; n < D_STATE; ++n) hs[n] = 0.0f;
    __shared__ float sB[D_STATE], sC[D_STATE];
    float Dh = Dp[h];

    for (int t = 0; t < SEQ; ++t) {
        size_t m = (size_t)b * SEQ + t;
        __syncthreads();
        if (p < D_STATE)            sB[p]           = xc[m * CONV_DIM + D_INNER + p];
        else if (p < 2 * D_STATE)   sC[p - D_STATE] = xc[m * CONV_DIM + D_INNER + p];
        __syncthreads();
        float dtv = dt[m * NHEADS + h];
        float dav = dA[m * NHEADS + h];
        float xv  = xc[m * CONV_DIM + h * HEADDIM + p];
        float dtx = dtv * xv;
        float y   = Dh * xv;
#pragma unroll
        for (int n = 0; n < D_STATE; ++n) {
            hs[n] = hs[n] * dav + dtx * sB[n];
            y += hs[n] * sC[n];
        }
        ys[m * D_INNER + h * HEADDIM + p] = y;
    }
}

// ---------------- y *= silu(z); RMSNorm; * norm_w; -> bf16 -------------------
__global__ __launch_bounds__(256)
void gate_rms(const float* __restrict__ ys, const float* __restrict__ zx,
              const float* __restrict__ nw, unsigned short* __restrict__ ybf)
{
    int m = blockIdx.x;
    int tid = threadIdx.x;
    __shared__ float red[256];
    float v[4];
    float ss = 0.0f;
#pragma unroll
    for (int j = 0; j < 4; ++j) {
        int c = tid + j * 256;
        float z = zx[(size_t)m * D_IN_PROJ + c];
        float g = ys[(size_t)m * D_INNER + c] * (z / (1.0f + __expf(-z)));
        v[j] = g;
        ss += g * g;
    }
    red[tid] = ss; __syncthreads();
    for (int s = 128; s > 0; s >>= 1) {
        if (tid < s) red[tid] += red[tid + s];
        __syncthreads();
    }
    float rstd = rsqrtf(red[0] / (float)D_INNER + EPS);
#pragma unroll
    for (int j = 0; j < 4; ++j) {
        int c = tid + j * 256;
        ybf[(size_t)m * D_INNER + c] = f32_to_bf16_rne(v[j] * rstd * nw[c]);
    }
}

// ---------------- h = x + xout + proj_b; LayerNorm -> out --------------------
__global__ __launch_bounds__(256)
void final_ln(const float* __restrict__ x, const float* __restrict__ xout,
              const float* __restrict__ pb, const float* __restrict__ g,
              const float* __restrict__ be, float* __restrict__ out)
{
    int m = blockIdx.x;
    int tid = threadIdx.x;
    __shared__ float r1[256], r2[256];
    int c0 = tid, c1 = tid + 256;
    float h0 = x[(size_t)m * D_MODEL + c0] + xout[(size_t)m * D_MODEL + c0] + pb[c0];
    float h1 = x[(size_t)m * D_MODEL + c1] + xout[(size_t)m * D_MODEL + c1] + pb[c1];
    r1[tid] = h0 + h1;
    r2[tid] = h0 * h0 + h1 * h1;
    __syncthreads();
    for (int s = 128; s > 0; s >>= 1) {
        if (tid < s) { r1[tid] += r1[tid + s]; r2[tid] += r2[tid + s]; }
        __syncthreads();
    }
    float mu   = r1[0] / (float)D_MODEL;
    float var  = r2[0] / (float)D_MODEL - mu * mu;
    float rstd = rsqrtf(var + EPS);
    out[(size_t)m * D_MODEL + c0] = (h0 - mu) * rstd * g[c0] + be[c0];
    out[(size_t)m * D_MODEL + c1] = (h1 - mu) * rstd * g[c1] + be[c1];
}

extern "C" void kernel_launch(void* const* d_in, const int* in_sizes, int n_in,
                              void* d_out, int out_size, void* d_ws, size_t ws_size,
                              hipStream_t stream)
{
    (void)in_sizes; (void)n_in; (void)out_size; (void)ws_size;
    const float* x          = (const float*)d_in[0];
    const float* W_in[2]    = {(const float*)d_in[1],  (const float*)d_in[9]};
    const float* conv_w[2]  = {(const float*)d_in[2],  (const float*)d_in[10]};
    const float* conv_b[2]  = {(const float*)d_in[3],  (const float*)d_in[11]};
    const float* dt_bias[2] = {(const float*)d_in[4],  (const float*)d_in[12]};
    const float* A_log[2]   = {(const float*)d_in[5],  (const float*)d_in[13]};
    const float* Dp[2]      = {(const float*)d_in[6],  (const float*)d_in[14]};
    const float* norm_w[2]  = {(const float*)d_in[7],  (const float*)d_in[15]};
    const float* W_out[2]   = {(const float*)d_in[8],  (const float*)d_in[16]};
    const float* proj_W     = (const float*)d_in[17];
    const float* proj_b     = (const float*)d_in[18];
    const float* ln_g       = (const float*)d_in[19];
    const float* ln_b       = (const float*)d_in[20];

    char* p = (char*)d_ws;
    auto carve = [&](size_t bytes) {
        char* r = p;
        p += (bytes + 255) & ~(size_t)255;
        return (void*)r;
    };
    // persistent
    unsigned short* xbf  = (unsigned short*)carve((size_t)MROWS * D_MODEL * 2);
    unsigned short* pwbf = (unsigned short*)carve((size_t)D_MODEL * D_INNER * 2);
    float*          ccf  = (float*)carve((size_t)MROWS * D_INNER * 4);
    unsigned short* ccbf = (unsigned short*)carve((size_t)MROWS * D_INNER * 2);
    float*          xout = (float*)carve((size_t)MROWS * D_MODEL * 4);
    // per-direction (reused)
    unsigned short* wibf = (unsigned short*)carve((size_t)D_IN_PROJ * D_MODEL * 2);
    unsigned short* wobf = (unsigned short*)carve((size_t)D_MODEL * D_INNER * 2);
    float*          zx   = (float*)carve((size_t)MROWS * D_IN_PROJ * 4);
    float*          xc   = (float*)carve((size_t)MROWS * CONV_DIM * 4);
    float*          dtb  = (float*)carve((size_t)MROWS * NHEADS * 4);
    float*          dab  = (float*)carve((size_t)MROWS * NHEADS * 4);
    float*          ys   = (float*)carve((size_t)MROWS * D_INNER * 4);
    unsigned short* ybf  = (unsigned short*)carve((size_t)MROWS * D_INNER * 2);

    int n;
    n = MROWS * D_MODEL;
    cvt_f32_bf16<<<(n + 255) / 256, 256, 0, stream>>>(x, xbf, n);
    n = D_MODEL * D_INNER;      // proj_W is (1024,512): K=1024, N=512
    transpose_cvt<<<(n + 255) / 256, 256, 0, stream>>>(proj_W, pwbf, D_INNER, D_MODEL);

    for (int d = 0; d < 2; ++d) {
        n = D_IN_PROJ * D_MODEL;  // W_in (512,2096): K=512, N=2096
        transpose_cvt<<<(n + 255) / 256, 256, 0, stream>>>(W_in[d], wibf, D_MODEL, D_IN_PROJ);

        dim3 g1(MROWS / 16, (D_IN_PROJ / 16 + 3) / 4);
        gemm_bf16<<<g1, 32, 0, stream>>>(xbf, wibf, zx,
                                         D_IN_PROJ, D_MODEL, D_IN_PROJ, 0,
                                         /*revA=*/d, /*revC=*/0, SEQ);

        n = MROWS * CONV_DIM;
        conv_silu<<<(n + 255) / 256, 256, 0, stream>>>(zx, conv_w[d], conv_b[d], xc);

        n = MROWS * NHEADS;
        dt_da<<<(n + 255) / 256, 256, 0, stream>>>(zx, dt_bias[d], A_log[d], dtb, dab);

        ssm_scan<<<B_SZ * NHEADS, 64, 0, stream>>>(xc, dtb, dab, Dp[d], ys);

        gate_rms<<<MROWS, 256, 0, stream>>>(ys, zx, norm_w[d], ybf);

        n = D_MODEL * D_INNER;    // W_out (1024,512): K=1024, N=512
        transpose_cvt<<<(n + 255) / 256, 256, 0, stream>>>(W_out[d], wobf, D_INNER, D_MODEL);

        dim3 g2(MROWS / 16, (D_MODEL / 16 + 3) / 4);
        gemm_bf16<<<g2, 32, 0, stream>>>(ybf, wobf, ccf,
                                         D_MODEL, D_INNER, D_INNER, d * D_MODEL,
                                         /*revA=*/0, /*revC=*/d, SEQ);
    }

    n = MROWS * D_INNER;
    cvt_f32_bf16<<<(n + 255) / 256, 256, 0, stream>>>(ccf, ccbf, n);

    dim3 g3(MROWS / 16, (D_MODEL / 16 + 3) / 4);
    gemm_bf16<<<g3, 32, 0, stream>>>(ccbf, pwbf, xout,
                                     D_MODEL, D_INNER, D_MODEL, 0, 0, 0, SEQ);

    final_ln<<<MROWS, 256, 0, stream>>>(x, xout, proj_b, ln_g, ln_b, (float*)d_out);
}
